// Net_89361089561102
// MI455X (gfx1250) — matrix operand order repaired and checked by reading, not compiled, not run on hardware
//
#include <hip/hip_runtime.h>
#include <hip/hip_bf16.h>
#include <math.h>

// ---------------- model constants ----------------
#define HDIM   3072
#define NHEAD  32
#define HEADD  96
#define FFDIM  8192
#define NLAYER 2
#define BATCH  2
#define SEQ    512
#define NTOK   (BATCH * SEQ)
#define EPSV   1e-5f

typedef _Float16 v8h  __attribute__((ext_vector_type(8)));
typedef _Float16 v16h __attribute__((ext_vector_type(16)));
typedef float    v8f  __attribute__((ext_vector_type(8)));

__constant__ float NF4LUT[16] = {
    -1.0f, -0.6961928009986877f, -0.5250730514526367f, -0.39491748809814453f,
    -0.28444138169288635f, -0.18477343022823334f, -0.09105003625154495f, 0.0f,
    0.07958029955625534f, 0.16093020141124725f, 0.24611230194568634f, 0.33791524171829224f,
    0.44070982933044434f, 0.5626170039176941f, 0.7229568362236023f, 1.0f};

// ---------------- helpers ----------------
__device__ __forceinline__ float block_sum(float v, float* red) {
  __syncthreads();
  for (int off = 16; off; off >>= 1) v += __shfl_xor(v, off, 32);
  const int lane = threadIdx.x & 31, w = threadIdx.x >> 5;
  if (lane == 0) red[w] = v;
  __syncthreads();
  if (w == 0) {
    float t = (lane < 8) ? red[lane] : 0.0f;
    for (int off = 4; off; off >>= 1) t += __shfl_xor(t, off, 32);
    if (lane == 0) red[0] = t;
  }
  __syncthreads();
  return red[0];
}

__device__ __forceinline__ v8f wmma_f16(v16h a, v16h b, v8f c) {
  // D = A(16x32 f16) * B(32x16 f16) + C(16x16 f32)
  return __builtin_amdgcn_wmma_f32_16x16x32_f16(false, a, false, b, (short)0, c,
                                                false, false);
}

// A-fragment in this code's contiguous k order (B uses the same): lane<16 holds row m0+lane, K 0..15;
// lane>=16 holds row m0+lane-16, K 16..31. Both halves contiguous 16B.
__device__ __forceinline__ v16h load_a_frag(const _Float16* p) {
  v8h lo = *(const v8h*)(p);
  v8h hi = *(const v8h*)(p + 8);
  return __builtin_shufflevector(lo, hi, 0, 1, 2, 3, 4, 5, 6, 7, 8, 9, 10, 11, 12, 13, 14, 15);
}

// ---------------- NF4 dequant: packed int32 -> f16 row-major [out_f, in_f] ----------------
__global__ __launch_bounds__(256) void dequant_nf4(const int* __restrict__ packed,
                                                   const float* __restrict__ absmax,
                                                   _Float16* __restrict__ out, int n_packed) {
  int i = blockIdx.x * 256 + threadIdx.x;
  if (i >= n_packed) return;
  unsigned u = (unsigned)packed[i];
  size_t base = (size_t)i * 8;
  float am = absmax[i >> 3];  // 64 weights per absmax block = 8 packed ints
  v8h o;
#pragma unroll
  for (int j = 0; j < 8; ++j) o[j] = (_Float16)(NF4LUT[(u >> (4 * j)) & 0xF] * am);
  *(v8h*)(out + base) = o;
}

// ---------------- embedding gather ----------------
__global__ __launch_bounds__(256) void embed_gather(const int* __restrict__ ids,
                                                    const float* __restrict__ embed,
                                                    float* __restrict__ x) {
  const int t = blockIdx.x;
  const size_t src = (size_t)ids[t] * HDIM;
  for (int i = threadIdx.x; i < HDIM; i += 256) x[(size_t)t * HDIM + i] = embed[src + i];
}

// ---------------- RMSNorm -> f16 ----------------
__global__ __launch_bounds__(256) void rmsnorm_f16(const float* __restrict__ x,
                                                   const float* __restrict__ w,
                                                   _Float16* __restrict__ out) {
  __shared__ float red[8];
  const size_t row = (size_t)blockIdx.x * HDIM;
  float ss = 0.0f;
  for (int i = threadIdx.x; i < HDIM; i += 256) {
    float v = x[row + i];
    ss += v * v;
  }
  ss = block_sum(ss, red);
  const float sc = rsqrtf(ss * (1.0f / HDIM) + EPSV);
  for (int i = threadIdx.x; i < HDIM; i += 256)
    out[row + i] = (_Float16)(x[row + i] * sc * w[i]);
}

// ---------------- WMMA GEMM: C[M,N] = A[M,K] * B[N,K]^T (+resid) ----------------
// block = 256 threads (8 waves); wave -> 32x64 C tile (2x4 frags); block -> 128x128.
__global__ __launch_bounds__(256) void gemm_nt_f16f32(const _Float16* __restrict__ A,
                                                      const _Float16* __restrict__ B,
                                                      const float* __restrict__ resid,
                                                      float* __restrict__ C, int M, int N,
                                                      int K) {
  const int lane = threadIdx.x & 31;
  const int wave = threadIdx.x >> 5;
  const int m0 = blockIdx.y * 128 + (wave & 3) * 32;
  const int n0 = blockIdx.x * 128 + (wave >> 2) * 64;

  const _Float16* pa0 = A + (size_t)(m0 + (lane & 15)) * K + ((lane >> 4) << 4);
  const _Float16* pa1 = pa0 + (size_t)16 * K;
  // B-fragment: lane<16 -> col n+lane, K 0..15 contiguous; lane>=16 -> K 16..31.
  const _Float16* pb0 = B + (size_t)(n0 + (lane & 15)) * K + ((lane >> 4) << 4);
  const _Float16* pb1 = pb0 + (size_t)16 * K;
  const _Float16* pb2 = pb0 + (size_t)32 * K;
  const _Float16* pb3 = pb0 + (size_t)48 * K;

  v8f a00 = {}, a01 = {}, a02 = {}, a03 = {};
  v8f a10 = {}, a11 = {}, a12 = {}, a13 = {};
#pragma unroll 2
  for (int k = 0; k < K; k += 32) {
    v16h f0 = load_a_frag(pa0 + k);
    v16h f1 = load_a_frag(pa1 + k);
    v16h b0 = *(const v16h*)(pb0 + k);
    v16h b1 = *(const v16h*)(pb1 + k);
    v16h b2 = *(const v16h*)(pb2 + k);
    v16h b3 = *(const v16h*)(pb3 + k);
    if (k + 128 < K) {
      __builtin_prefetch(pa0 + k + 128, 0, 1);
      __builtin_prefetch(pa1 + k + 128, 0, 1);
      __builtin_prefetch(pb0 + k + 128, 0, 1);
      __builtin_prefetch(pb1 + k + 128, 0, 1);
      __builtin_prefetch(pb2 + k + 128, 0, 1);
      __builtin_prefetch(pb3 + k + 128, 0, 1);
    }
    a00 = wmma_f16(f0, b0, a00);
    a01 = wmma_f16(f0, b1, a01);
    a02 = wmma_f16(f0, b2, a02);
    a03 = wmma_f16(f0, b3, a03);
    a10 = wmma_f16(f1, b0, a10);
    a11 = wmma_f16(f1, b1, a11);
    a12 = wmma_f16(f1, b2, a12);
    a13 = wmma_f16(f1, b3, a13);
  }

  // C/D layout: VGPR r -> row (m0+r) lanes 0-15 / (m0+r+8) lanes 16-31, col n0+(lane&15)
  const int rbase = (lane >> 4) << 3;
  const int cbase = lane & 15;
  v8f accs[2][4] = {{a00, a01, a02, a03}, {a10, a11, a12, a13}};
#pragma unroll
  for (int mi = 0; mi < 2; ++mi) {
#pragma unroll
    for (int nj = 0; nj < 4; ++nj) {
#pragma unroll
      for (int r = 0; r < 8; ++r) {
        const int row = m0 + mi * 16 + rbase + r;
        const size_t idx = (size_t)row * N + n0 + nj * 16 + cbase;
        float v = accs[mi][nj][r];
        if (resid) v += resid[idx];
        C[idx] = v;
      }
    }
  }
}

// ---------------- RoPE + pack q/k into [BH,S,HD], v transposed into [BH,HD,S] ----------------
__global__ __launch_bounds__(256) void rope_pack(const float* __restrict__ qkv,
                                                 _Float16* __restrict__ q16,
                                                 _Float16* __restrict__ k16,
                                                 _Float16* __restrict__ v16t) {
  const int t = blockIdx.x;
  const int b = t / SEQ, s = t - b * SEQ;
  const float qscale = 0.10206207261596577f;  // 1/sqrt(HEADD)
  const float* base = qkv + (size_t)t * (3 * HDIM);
  for (int i = threadIdx.x; i < 3 * HDIM; i += 256) {
    const int which = i / HDIM;
    const int r = i - which * HDIM;
    const int hh = r / HEADD;
    const int d = r - hh * HEADD;
    const float val = base[i];
    const int bh = b * NHEAD + hh;
    if (which == 2) {  // V stored transposed: [BH][HEADD][SEQ]
      v16t[((size_t)bh * HEADD + d) * SEQ + s] = (_Float16)val;
      continue;
    }
    const size_t dst = ((size_t)bh * SEQ + s) * HEADD + d;
    const int dj = (d < 48) ? d : d - 48;
    const float ang = (float)s * __powf(10000.0f, -(float)dj * (1.0f / 48.0f));
    float sn, cs;
    __sincosf(ang, &sn, &cs);
    const int d2 = (d < 48) ? d + 48 : d - 48;
    const float part = base[which * HDIM + hh * HEADD + d2];
    const float rot = (d < 48) ? -part : part;
    const float o = val * cs + rot * sn;
    if (which == 0)
      q16[dst] = (_Float16)(o * qscale);
    else
      k16[dst] = (_Float16)o;
  }
}

// ---------------- flash attention (WMMA): wave = 16 q-rows; block = 4 waves = 64 q-rows ----------------
__global__ __launch_bounds__(128) void flash_attn(const _Float16* __restrict__ q,
                                                  const _Float16* __restrict__ k,
                                                  const _Float16* __restrict__ vt,
                                                  const int* __restrict__ amask,
                                                  _Float16* __restrict__ ctx) {
  const int bh = blockIdx.y;  // b*NHEAD + h
  const int b = bh / NHEAD;
  const int h = bh - b * NHEAD;
  const int lane = threadIdx.x & 31;
  const int wave = threadIdx.x >> 5;  // 0..3
  const int qt0 = blockIdx.x * 64 + wave * 16;

  const _Float16* qh = q + (size_t)bh * SEQ * HEADD;
  const _Float16* kh = k + (size_t)bh * SEQ * HEADD;
  const _Float16* vh = vt + (size_t)bh * HEADD * SEQ;

  __shared__ __align__(32) _Float16 pbuf[4][16][32];  // per-wave P tile (C-layout -> A-frag relay)

  // Q A-fragments for the 3 K-dim steps (d = 0, 32, 64)
  const _Float16* qp = qh + (size_t)(qt0 + (lane & 15)) * HEADD + ((lane >> 4) << 4);
  const v16h aq0 = load_a_frag(qp);
  const v16h aq1 = load_a_frag(qp + 32);
  const v16h aq2 = load_a_frag(qp + 64);

  v8f cacc[6];
#pragma unroll
  for (int f = 0; f < 6; ++f) cacc[f] = (v8f){};
  float mrow[8], lrow[8];
#pragma unroll
  for (int r = 0; r < 8; ++r) {
    mrow[r] = -1e30f;
    lrow[r] = 0.0f;
  }

  const int rbase = (lane >> 4) << 3;
  const int cidx = lane & 15;

  for (int kt = 0; kt < qt0 + 16; kt += 32) {
    // ---- scores for 32 keys: two 16-key groups, 3 wmma each (K=96) ----
    v8f s[2];
#pragma unroll
    for (int g = 0; g < 2; ++g) {
      const int key = kt + g * 16 + cidx;
      const _Float16* kp = kh + (size_t)key * HEADD + ((lane >> 4) << 4);
      v16h b0 = *(const v16h*)(kp);
      v16h b1 = *(const v16h*)(kp + 32);
      v16h b2 = *(const v16h*)(kp + 64);
      v8f acc = {};
      acc = wmma_f16(aq0, b0, acc);
      acc = wmma_f16(aq1, b1, acc);
      acc = wmma_f16(aq2, b2, acc);
      const bool mok = amask[b * SEQ + key] != 0;
#pragma unroll
      for (int r = 0; r < 8; ++r) {
        const int row = qt0 + rbase + r;
        if (key > row || !mok) acc[r] = -1e30f;
      }
      s[g] = acc;
    }
    // ---- online softmax (row stats live within one 16-lane half) ----
#pragma unroll
    for (int r = 0; r < 8; ++r) {
      float t = fmaxf(s[0][r], s[1][r]);
      for (int off = 8; off; off >>= 1) t = fmaxf(t, __shfl_xor(t, off, 32));
      const float mnew = fmaxf(mrow[r], t);
      const float alpha = __expf(mrow[r] - mnew);
      const float p0 = __expf(s[0][r] - mnew);
      const float p1 = __expf(s[1][r] - mnew);
      float ts = p0 + p1;
      for (int off = 8; off; off >>= 1) ts += __shfl_xor(ts, off, 32);
      lrow[r] = lrow[r] * alpha + ts;
      mrow[r] = mnew;
#pragma unroll
      for (int f = 0; f < 6; ++f) cacc[f][r] *= alpha;
      // stash P (f16) into per-wave LDS tile, row-major 16x32
      const int prow = rbase + r;
      pbuf[wave][prow][cidx] = (_Float16)p0;
      pbuf[wave][prow][cidx + 16] = (_Float16)p1;
    }
    // ---- reload P as an A-fragment (16 rows x 32 keys); wave-local LDS is in-order ----
    const _Float16* pp = &pbuf[wave][lane & 15][(lane >> 4) << 4];
    const v16h pa = load_a_frag(pp);
    // ---- ctx += P * V  (V transposed: B-frag cols are contiguous key runs) ----
#pragma unroll
    for (int f = 0; f < 6; ++f) {
      const int d = f * 16 + cidx;
      const _Float16* vp = vh + (size_t)d * SEQ + kt + ((lane >> 4) << 4);
      const v16h bv = *(const v16h*)vp;
      cacc[f] = wmma_f16(pa, bv, cacc[f]);
    }
  }

  // ---- normalize + store ctx in token-major [B*S, H] ----
#pragma unroll
  for (int r = 0; r < 8; ++r) {
    const int row = qt0 + rbase + r;
    const float inv = 1.0f / lrow[r];
#pragma unroll
    for (int f = 0; f < 6; ++f) {
      const int d = f * 16 + cidx;
      ctx[((size_t)(b * SEQ + row)) * HDIM + h * HEADD + d] = (_Float16)(cacc[f][r] * inv);
    }
  }
}

// ---------------- SiLU gate: act = up * silu(gate) ----------------
__global__ __launch_bounds__(256) void silu_mul(const float* __restrict__ gu,
                                                _Float16* __restrict__ act, int total) {
  int i = blockIdx.x * 256 + threadIdx.x;
  if (i >= total) return;
  const int row = i / FFDIM;
  const int col = i - row * FFDIM;
  const float g = gu[(size_t)row * (2 * FFDIM) + col];
  const float u = gu[(size_t)row * (2 * FFDIM) + FFDIM + col];
  act[i] = (_Float16)(u * g / (1.0f + __expf(-g)));
}

// ---------------- final RMSNorm + cls head on last token only ----------------
__global__ __launch_bounds__(256) void head_kernel(const float* __restrict__ x,
                                                   const float* __restrict__ fln,
                                                   const float* __restrict__ w1,
                                                   const float* __restrict__ b1,
                                                   const float* __restrict__ lng,
                                                   const float* __restrict__ lnb,
                                                   const float* __restrict__ w2,
                                                   const float* __restrict__ b2,
                                                   float* __restrict__ out) {
  const int b = blockIdx.x;
  const float* row = x + ((size_t)b * SEQ + (SEQ - 1)) * HDIM;
  __shared__ float xr[HDIM];
  __shared__ float hbuf[768];
  __shared__ float red[8];
  const int tid = threadIdx.x;

  float ss = 0.0f;
  for (int i = tid; i < HDIM; i += 256) {
    const float v = row[i];
    xr[i] = v;
    ss += v * v;
  }
  ss = block_sum(ss, red);
  const float sc = rsqrtf(ss * (1.0f / HDIM) + EPSV);
  for (int i = tid; i < HDIM; i += 256) xr[i] = xr[i] * sc * fln[i];
  __syncthreads();

  for (int j = tid; j < 768; j += 256) {
    float d = b1[j];
    const float* wr = w1 + (size_t)j * HDIM;
    for (int i = 0; i < HDIM; ++i) d += xr[i] * wr[i];
    hbuf[j] = fmaxf(d, 0.0f);
  }
  __syncthreads();

  float sm = 0.0f, sq = 0.0f;
  for (int j = tid; j < 768; j += 256) {
    const float v = hbuf[j];
    sm += v;
    sq += v * v;
  }
  sm = block_sum(sm, red);
  sq = block_sum(sq, red);
  const float mu = sm * (1.0f / 768.0f);
  const float var = sq * (1.0f / 768.0f) - mu * mu;
  const float inv = rsqrtf(var + EPSV);
  for (int j = tid; j < 768; j += 256) hbuf[j] = (hbuf[j] - mu) * inv * lng[j] + lnb[j];
  __syncthreads();

  for (int c = 0; c < 2; ++c) {
    float d = 0.0f;
    for (int j = tid; j < 768; j += 256) d += hbuf[j] * w2[c * 768 + j];
    d = block_sum(d, red);
    if (tid == 0) out[b * 2 + c] = d + b2[c];
  }
}

// ---------------- host orchestration ----------------
extern "C" void kernel_launch(void* const* d_in, const int* in_sizes, int n_in, void* d_out,
                              int out_size, void* d_ws, size_t ws_size, hipStream_t stream) {
  (void)in_sizes;
  (void)n_in;
  (void)out_size;
  (void)ws_size;
  const int* ids = (const int*)d_in[0];
  const int* amask = (const int*)d_in[1];
  const float* embed = (const float*)d_in[2];
  const int* qkvP = (const int*)d_in[3];
  const float* qkvA = (const float*)d_in[4];
  const int* oP = (const int*)d_in[5];
  const float* oA = (const float*)d_in[6];
  const int* guP = (const int*)d_in[7];
  const float* guA = (const float*)d_in[8];
  const int* dnP = (const int*)d_in[9];
  const float* dnA = (const float*)d_in[10];
  const float* ln1 = (const float*)d_in[11];
  const float* ln2 = (const float*)d_in[12];
  const float* fln = (const float*)d_in[13];
  const float* w1 = (const float*)d_in[14];
  const float* b1 = (const float*)d_in[15];
  const float* lng = (const float*)d_in[16];
  const float* lnb = (const float*)d_in[17];
  const float* w2 = (const float*)d_in[18];
  const float* b2 = (const float*)d_in[19];

  const size_t NQKV = (size_t)3 * HDIM * HDIM;
  const size_t NO = (size_t)HDIM * HDIM;
  const size_t NGU = (size_t)2 * FFDIM * HDIM;
  const size_t ND = (size_t)HDIM * FFDIM;

  char* ws = (char*)d_ws;
  size_t off = 0;
  auto take = [&](size_t bytes) -> void* {
    void* p = ws + off;
    off = (off + bytes + 255) & ~(size_t)255;
    return p;
  };
  float* x = (float*)take((size_t)NTOK * HDIM * 4);          // residual stream
  _Float16* h16 = (_Float16*)take((size_t)NTOK * HDIM * 2);  // normed activations
  _Float16* wdeq = (_Float16*)take(NGU * 2);                 // dequantized weights
  float* big = (float*)take((size_t)NTOK * 2 * FFDIM * 4);   // qkv / gate-up output
  _Float16* q16 = (_Float16*)take((size_t)NTOK * HDIM * 2);
  _Float16* k16 = (_Float16*)take((size_t)NTOK * HDIM * 2);
  _Float16* v16t = (_Float16*)take((size_t)NTOK * HDIM * 2);
  _Float16* a16 = (_Float16*)take((size_t)NTOK * FFDIM * 2);  // ctx / mlp act

  auto gemm = [&](const _Float16* A, const _Float16* Bw, const float* R, float* C, int M,
                  int N, int K) {
    dim3 grid(N / 128, M / 128);
    gemm_nt_f16f32<<<grid, 256, 0, stream>>>(A, Bw, R, C, M, N, K);
  };
  auto dequant = [&](const int* p, const float* a, size_t n) {
    int np = (int)(n / 8);
    dequant_nf4<<<(np + 255) / 256, 256, 0, stream>>>(p, a, wdeq, np);
  };

  embed_gather<<<NTOK, 256, 0, stream>>>(ids, embed, x);

  for (int l = 0; l < NLAYER; ++l) {
    // ---- attention ----
    rmsnorm_f16<<<NTOK, 256, 0, stream>>>(x, ln1 + (size_t)l * HDIM, h16);
    dequant(qkvP + (size_t)l * (NQKV / 8), qkvA + (size_t)l * (NQKV / 64), NQKV);
    gemm(h16, wdeq, nullptr, big, NTOK, 3 * HDIM, HDIM);
    rope_pack<<<NTOK, 256, 0, stream>>>(big, q16, k16, v16t);
    {
      dim3 ag(SEQ / 64, BATCH * NHEAD);
      flash_attn<<<ag, 128, 0, stream>>>(q16, k16, v16t, amask, a16);
    }
    dequant(oP + (size_t)l * (NO / 8), oA + (size_t)l * (NO / 64), NO);
    gemm(a16, wdeq, x, x, NTOK, HDIM, HDIM);  // x += ctx @ Wo^T
    // ---- gated MLP ----
    rmsnorm_f16<<<NTOK, 256, 0, stream>>>(x, ln2 + (size_t)l * HDIM, h16);
    dequant(guP + (size_t)l * (NGU / 8), guA + (size_t)l * (NGU / 64), NGU);
    gemm(h16, wdeq, nullptr, big, NTOK, 2 * FFDIM, HDIM);
    {
      int total = NTOK * FFDIM;
      silu_mul<<<(total + 255) / 256, 256, 0, stream>>>(big, a16, total);
    }
    dequant(dnP + (size_t)l * (ND / 8), dnA + (size_t)l * (ND / 64), ND);
    gemm(a16, wdeq, x, x, NTOK, HDIM, FFDIM);  // x += act @ Wd^T
  }

  head_kernel<<<BATCH, 256, 0, stream>>>(x, fln, w1, b1, lng, lnb, w2, b2, (float*)d_out);
}
